// GINEEncoder_2800318677654
// MI455X (gfx1250) — compile-verified, matrix operand order checked by reading, not run on hardware
//
#include <hip/hip_runtime.h>
#include <cstdint>

#define N_GRAPHS 1024
#define NODES_PER_GRAPH 32
#define NN (N_GRAPHS*NODES_PER_GRAPH)   // 32768 nodes
#define NE 131072
#define HID 256
#define OUTDIM 512
#define NL 5

typedef _Float16 half_t;
typedef __attribute__((ext_vector_type(16))) _Float16 v16h;
typedef __attribute__((ext_vector_type(8)))  _Float16 v8h;
typedef __attribute__((ext_vector_type(4)))  _Float16 v4h;
typedef __attribute__((ext_vector_type(8)))  float    v8f;
typedef __attribute__((ext_vector_type(4)))  float    v4f;

enum { MODE_RELU_F16 = 0, MODE_BN_RELU_F32 = 1, MODE_BN_F32 = 2, MODE_BIAS_F32 = 3 };

// ---------------- elementwise / gather-scatter kernels ----------------

__global__ void k_zero4(float* __restrict__ p) {
    int t = blockIdx.x * 256 + threadIdx.x;
    v4f z = {0.f, 0.f, 0.f, 0.f};
    ((v4f*)p)[t] = z;
}

// h[node] = sum_f atom_tables[f][x[node,f]]   (tables ~1 MB, L2 resident)
__global__ void k_embed_atoms(const int* __restrict__ x,
                              const float* __restrict__ tables,
                              float* __restrict__ h) {
    int t = blockIdx.x * 256 + threadIdx.x;   // NN*64 threads
    int node = t >> 6;
    int c = (t & 63) << 2;
    v4f acc = {0.f, 0.f, 0.f, 0.f};
    #pragma unroll
    for (int f = 0; f < 9; ++f) {
        int idx = x[node * 9 + f];
        v4f v = *(const v4f*)(tables + ((size_t)f * 119 + idx) * HID + c);
        acc[0] += v[0]; acc[1] += v[1]; acc[2] += v[2]; acc[3] += v[3];
    }
    *(v4f*)(h + (size_t)node * HID + c) = acc;
}

// aggr[dst] += relu(h[src] + bond_embed(e))  — bond tables are 18 KB, recomputed inline
__global__ void k_msg(const int* __restrict__ ei,
                      const int* __restrict__ eattr,
                      const float* __restrict__ btab,
                      const float* __restrict__ h,
                      float* __restrict__ aggr) {
    int t = blockIdx.x * 256 + threadIdx.x;   // NE*64 threads
    int e = t >> 6;
    int c = (t & 63) << 2;
    int src = ei[e];
    int dst = ei[NE + e];
    v4f m = *(const v4f*)(h + (size_t)src * HID + c);
    #pragma unroll
    for (int f = 0; f < 3; ++f) {
        int idx = eattr[e * 3 + f];
        v4f v = *(const v4f*)(btab + ((size_t)f * 6 + idx) * HID + c);
        m[0] += v[0]; m[1] += v[1]; m[2] += v[2]; m[3] += v[3];
    }
    float* out = aggr + (size_t)dst * HID + c;
    atomicAdd(out + 0, fmaxf(m[0], 0.f));
    atomicAdd(out + 1, fmaxf(m[1], 0.f));
    atomicAdd(out + 2, fmaxf(m[2], 0.f));
    atomicAdd(out + 3, fmaxf(m[3], 0.f));
}

// z = (1+eps)*h + aggr  -> f16 (A-matrix input of GEMM1)
__global__ void k_make_z(const float* __restrict__ h, const float* __restrict__ aggr,
                         const float* __restrict__ eps, int l,
                         half_t* __restrict__ z) {
    int t = blockIdx.x * 256 + threadIdx.x;   // NN*64 threads
    int node = t >> 6;
    int c = (t & 63) << 2;
    float s = 1.0f + eps[l];
    v4f hv = *(const v4f*)(h + (size_t)node * HID + c);
    v4f av = *(const v4f*)(aggr + (size_t)node * HID + c);
    v4h o;
    o[0] = (half_t)(s * hv[0] + av[0]);
    o[1] = (half_t)(s * hv[1] + av[1]);
    o[2] = (half_t)(s * hv[2] + av[2]);
    o[3] = (half_t)(s * hv[3] + av[3]);
    *(v4h*)(z + (size_t)node * HID + c) = o;
}

// out[l][n][k] = (f16) in[l][k][n]   — weight transpose+convert, B^T layout [N,K]
__global__ void k_transpose_to_f16(const float* __restrict__ in, half_t* __restrict__ out,
                                   int K, int N, int total) {
    int t = blockIdx.x * 256 + threadIdx.x;
    if (t >= total) return;
    int kn = K * N;
    int l = t / kn;
    int rem = t - l * kn;
    int n = rem / K;
    int k = rem - n * K;
    out[t] = (half_t)in[(size_t)l * kn + (size_t)k * N + n];
}

// mean over 32 nodes/graph -> f16
__global__ void k_pool(const float* __restrict__ node_emb, half_t* __restrict__ gH) {
    int t = blockIdx.x * 256 + threadIdx.x;   // N_GRAPHS*64 threads
    int g = t >> 6;
    int c = (t & 63) << 2;
    v4f acc = {0.f, 0.f, 0.f, 0.f};
    const float* base = node_emb + (size_t)g * NODES_PER_GRAPH * HID + c;
    #pragma unroll 4
    for (int i = 0; i < NODES_PER_GRAPH; ++i) {
        v4f v = *(const v4f*)(base + (size_t)i * HID);
        acc[0] += v[0]; acc[1] += v[1]; acc[2] += v[2]; acc[3] += v[3];
    }
    v4h o;
    o[0] = (half_t)(acc[0] * (1.f / 32.f));
    o[1] = (half_t)(acc[1] * (1.f / 32.f));
    o[2] = (half_t)(acc[2] * (1.f / 32.f));
    o[3] = (half_t)(acc[3] * (1.f / 32.f));
    *(v4h*)(gH + (size_t)g * HID + c) = o;
}

__global__ void k_mask(float* __restrict__ m) {
    m[blockIdx.x * 256 + threadIdx.x] = 1.0f;
}

// ---------------- WMMA GEMM: C[M,N] = A[M,K] * BT[N,K]^T + bias (+BN)(+relu) ----------------
// wave32; each wave -> 32x64 output tile (8 f32 accumulators, B fragment reused 2x).
// 8 waves/block: 4 along M x 2 along N -> 128x128 block tile (B shared across waves via WGP$).
// A fragment: lane l holds row (l&15), K-chunk base (l>=16?8:0): halves [b..b+7],[b+16..b+23].
// B fragment: identical addressing on BT[N,K] (column-major consumption per ISA layout).
__global__ void k_wmma_gemm(const half_t* __restrict__ A,
                            const half_t* __restrict__ BT,
                            const float* __restrict__ bias,
                            const float* __restrict__ bn_g,
                            const float* __restrict__ bn_b,
                            const float* __restrict__ bn_m,
                            const float* __restrict__ bn_v,
                            half_t* __restrict__ outH,
                            float* __restrict__ outF,
                            int M, int K, int N, int mode) {
    const int lane = threadIdx.x & 31;
    const int wave = threadIdx.x >> 5;
    const int mTile = blockIdx.x * 128 + (wave & 3) * 32;
    const int nTile = blockIdx.y * 128 + (wave >> 2) * 64;
    const int hs = lane >> 4;        // K-half select
    const int lr = lane & 15;
    (void)M;

    v8f acc[2][4];
    for (int mi = 0; mi < 2; ++mi)
        for (int j = 0; j < 4; ++j)
            for (int i = 0; i < 8; ++i) acc[mi][j][i] = 0.f;

    const half_t* Arow0 = A + (size_t)(mTile + lr) * K + hs * 8;
    const half_t* Arow1 = Arow0 + (size_t)16 * K;
    const half_t* Brow  = BT + (size_t)(nTile + lr) * K + hs * 8;

    for (int k0 = 0; k0 < K; k0 += 32) {
        v16h a0, a1;
        {
            v8h lo = *(const v8h*)(Arow0 + k0);
            v8h hi = *(const v8h*)(Arow0 + k0 + 16);
            #pragma unroll
            for (int i = 0; i < 8; ++i) { a0[i] = lo[i]; a0[i + 8] = hi[i]; }
        }
        {
            v8h lo = *(const v8h*)(Arow1 + k0);
            v8h hi = *(const v8h*)(Arow1 + k0 + 16);
            #pragma unroll
            for (int i = 0; i < 8; ++i) { a1[i] = lo[i]; a1[i + 8] = hi[i]; }
        }
        #pragma unroll
        for (int j = 0; j < 4; ++j) {
            const half_t* bp = Brow + (size_t)j * 16 * K + k0;
            v8h blo = *(const v8h*)(bp);
            v8h bhi = *(const v8h*)(bp + 16);
            v16h b;
            #pragma unroll
            for (int i = 0; i < 8; ++i) { b[i] = blo[i]; b[i + 8] = bhi[i]; }
            acc[0][j] = __builtin_amdgcn_wmma_f32_16x16x32_f16(
                false, a0, false, b, (short)0, acc[0][j], false, false);
            acc[1][j] = __builtin_amdgcn_wmma_f32_16x16x32_f16(
                false, a1, false, b, (short)0, acc[1][j], false, false);
        }
    }

    // Epilogue. C/D layout: VGPR r, lanes 0-15 -> (M=r, N=lane); lanes 16-31 -> (M=8+r, N=lane-16)
    #pragma unroll
    for (int j = 0; j < 4; ++j) {
        const int n = nTile + j * 16 + lr;
        float bv = bias[n];
        float scale = 1.f, shift = 0.f;
        if (mode == MODE_BN_RELU_F32 || mode == MODE_BN_F32) {
            float sc = bn_g[n] * rsqrtf(bn_v[n] + 1e-5f);
            scale = sc;
            shift = bn_b[n] - bn_m[n] * sc;
        }
        #pragma unroll
        for (int mi = 0; mi < 2; ++mi) {
            const int rowBase = mTile + mi * 16 + hs * 8;
            #pragma unroll
            for (int r = 0; r < 8; ++r) {
                float v = (acc[mi][j][r] + bv) * scale + shift;
                if (mode == MODE_RELU_F16 || mode == MODE_BN_RELU_F32) v = fmaxf(v, 0.f);
                size_t off = (size_t)(rowBase + r) * N + n;
                if (mode == MODE_RELU_F16) outH[off] = (half_t)v;
                else                       outF[off] = v;
            }
        }
    }
}

// ---------------- launch ----------------

extern "C" void kernel_launch(void* const* d_in, const int* in_sizes, int n_in,
                              void* d_out, int out_size, void* d_ws, size_t ws_size,
                              hipStream_t stream) {
    (void)in_sizes; (void)n_in; (void)out_size; (void)ws_size;

    const int*   x     = (const int*)d_in[0];
    const int*   eattr = (const int*)d_in[1];
    const int*   ei    = (const int*)d_in[2];
    /* d_in[3] = batch: uniform sizes, unused */
    const float* atab  = (const float*)d_in[4];
    const float* btab  = (const float*)d_in[5];
    const float* eps   = (const float*)d_in[6];
    const float* w1    = (const float*)d_in[7];
    const float* b1    = (const float*)d_in[8];
    const float* w2    = (const float*)d_in[9];
    const float* b2    = (const float*)d_in[10];
    const float* bn_g  = (const float*)d_in[11];
    const float* bn_b  = (const float*)d_in[12];
    const float* bn_m  = (const float*)d_in[13];
    const float* bn_v  = (const float*)d_in[14];
    const float* pw1   = (const float*)d_in[15];
    const float* pb1   = (const float*)d_in[16];
    const float* pw2   = (const float*)d_in[17];
    const float* pb2   = (const float*)d_in[18];

    float* node_emb  = (float*)d_out;                       // [1024,32,256]
    float* node_mask = node_emb + (size_t)NN * HID;         // [1024,32]
    float* g_proj    = node_mask + NN;                      // [1024,512]

    uint8_t* ws = (uint8_t*)d_ws;
    size_t off = 0;
    auto alloc = [&](size_t bytes) -> void* {
        void* p = ws + off;
        off += (bytes + 255) & ~(size_t)255;
        return p;
    };
    float*  h    = (float*)alloc((size_t)NN * HID * 4);          // node features f32
    float*  aggr = (float*)alloc((size_t)NN * HID * 4);          // scatter-add target
    half_t* zH   = (half_t*)alloc((size_t)NN * HID * 2);         // GEMM1 A
    half_t* y1H  = (half_t*)alloc((size_t)NN * 2 * HID * 2);     // GEMM2 A
    half_t* w1T  = (half_t*)alloc((size_t)NL * 2 * HID * HID * 2);
    half_t* w2T  = (half_t*)alloc((size_t)NL * HID * 2 * HID * 2);
    half_t* p1T  = (half_t*)alloc((size_t)HID * HID * 2);
    half_t* p2T  = (half_t*)alloc((size_t)OUTDIM * HID * 2);
    half_t* gH   = (half_t*)alloc((size_t)N_GRAPHS * HID * 2);
    half_t* ghH  = (half_t*)alloc((size_t)N_GRAPHS * HID * 2);

    // --- weight prep: transpose + f16 convert (per call, deterministic) ---
    {
        int t1 = NL * HID * 2 * HID;
        k_transpose_to_f16<<<(t1 + 255) / 256, 256, 0, stream>>>(w1, w1T, HID, 2 * HID, t1);
        int t2 = NL * 2 * HID * HID;
        k_transpose_to_f16<<<(t2 + 255) / 256, 256, 0, stream>>>(w2, w2T, 2 * HID, HID, t2);
        int t3 = HID * HID;
        k_transpose_to_f16<<<(t3 + 255) / 256, 256, 0, stream>>>(pw1, p1T, HID, HID, t3);
        int t4 = HID * OUTDIM;
        k_transpose_to_f16<<<(t4 + 255) / 256, 256, 0, stream>>>(pw2, p2T, HID, OUTDIM, t4);
    }

    // --- atom embeddings ---
    k_embed_atoms<<<NN * 64 / 256, 256, 0, stream>>>(x, atab, h);

    // --- GINE layers ---
    for (int l = 0; l < NL; ++l) {
        k_zero4<<<(NN * HID / 4) / 256, 256, 0, stream>>>(aggr);
        k_msg<<<NE * 64 / 256, 256, 0, stream>>>(ei, eattr, btab, h, aggr);
        k_make_z<<<NN * 64 / 256, 256, 0, stream>>>(h, aggr, eps, l, zH);

        dim3 g1(NN / 128, (2 * HID) / 128);
        k_wmma_gemm<<<g1, 256, 0, stream>>>(zH, w1T + (size_t)l * 2 * HID * HID,
                                            b1 + (size_t)l * 2 * HID,
                                            nullptr, nullptr, nullptr, nullptr,
                                            y1H, nullptr, NN, HID, 2 * HID, MODE_RELU_F16);

        float* outp = (l == NL - 1) ? node_emb : h;
        int mode = (l == NL - 1) ? MODE_BN_F32 : MODE_BN_RELU_F32;
        dim3 g2(NN / 128, HID / 128);
        k_wmma_gemm<<<g2, 256, 0, stream>>>(y1H, w2T + (size_t)l * HID * 2 * HID,
                                            b2 + (size_t)l * HID,
                                            bn_g + (size_t)l * HID, bn_b + (size_t)l * HID,
                                            bn_m + (size_t)l * HID, bn_v + (size_t)l * HID,
                                            nullptr, outp, NN, 2 * HID, HID, mode);
    }

    // --- global mean pool + projection head ---
    k_pool<<<N_GRAPHS * 64 / 256, 256, 0, stream>>>(node_emb, gH);

    dim3 gp1(N_GRAPHS / 128, HID / 128);
    k_wmma_gemm<<<gp1, 256, 0, stream>>>(gH, p1T, pb1,
                                         nullptr, nullptr, nullptr, nullptr,
                                         ghH, nullptr, N_GRAPHS, HID, HID, MODE_RELU_F16);
    dim3 gp2(N_GRAPHS / 128, OUTDIM / 128);
    k_wmma_gemm<<<gp2, 256, 0, stream>>>(ghH, p2T, pb2,
                                         nullptr, nullptr, nullptr, nullptr,
                                         nullptr, g_proj, N_GRAPHS, HID, OUTDIM, MODE_BIAS_F32);

    k_mask<<<NN / 256, 256, 0, stream>>>(node_mask);
}
